// HunyuanImage3DecoderLayer_86775519248873
// MI455X (gfx1250) — matrix operand order repaired and checked by reading, not compiled, not run on hardware
//
// HunyuanImage3 decoder layer for MI455X (gfx1250, wave32, WMMA).
// All matmuls -> v_wmma_f32_16x16x32_bf16. fp32 weights stream HBM -> LDS via
// the Tensor Data Mover, double-buffered so the 23.3 TB/s weight stream (the
// binding resource: 1.8 GB of expert weights ~= 77us) overlaps WMMA compute.
// fp32 tiles are converted to packed bf16 in LDS cooperatively (8 elems per
// thread) so B-fragment builds are plain ds_load_b128, not per-lane cvt chains.
#include <hip/hip_runtime.h>
#include <cstdint>

// ---------------- problem constants ----------------
constexpr int SEQ   = 1024;
constexpr int HID   = 768;
constexpr int NHEADS = 16;
constexpr int KVHEADS = 8;
constexpr int HD    = 64;     // head dim
constexpr int GRP   = NHEADS / KVHEADS;  // 2
constexpr int HQD   = NHEADS * HD;       // 1024
constexpr int NEXP  = 64;
constexpr int TOPK  = 8;
constexpr int FFD   = 3072;
constexpr int CAPE  = 2 * SEQ * TOPK / NEXP; // 256
constexpr float EPSV = 1e-6f;
constexpr float F32EPS = 1.1920929e-07f;

#define CDIV(a,b) (((a)+(b)-1)/(b))

// ---------------- CDNA5 types ----------------
typedef __attribute__((ext_vector_type(16))) __bf16      v16bf;
typedef __attribute__((ext_vector_type(8)))  float       v8f;
typedef __attribute__((ext_vector_type(4)))  unsigned int u32x4;
typedef __attribute__((ext_vector_type(8)))  int          i32x8;
typedef __attribute__((ext_vector_type(4)))  int          i32x4;

union frag16 { v16bf v; unsigned short u[16]; uint4 q[2]; };

__device__ __forceinline__ unsigned short f2bf(float f) {
  union { float f; unsigned u; } x; x.f = f;
  unsigned r = x.u + 0x7fffu + ((x.u >> 16) & 1u);   // RNE
  return (unsigned short)(r >> 16);
}

// ---------------- Tensor Data Mover: 2D fp32 tile -> LDS ----------------
// D# per CDNA5 ISA (group0 128b, group1 256b).
// tile: tile_k elems (contiguous, dim0) x tile_n rows (dim1); row stride in
// elements; data_size=4B (code 2). Groups 2/3 zero (<=2D tensor).
__device__ __forceinline__ void tdm_load_tile_f32(
    unsigned lds_off, const float* gptr,
    unsigned tile_k, unsigned tile_n,
    unsigned tensor_k, unsigned tensor_n, unsigned row_stride) {
  unsigned long long ga = (unsigned long long)(uintptr_t)gptr;
  u32x4 g0;
  g0[0] = 1u;                                   // count=1, user descriptor
  g0[1] = lds_off;                              // lds_addr [63:32]
  g0[2] = (unsigned)(ga & 0xffffffffu);         // global_addr[31:0]  -> [95:64]
  g0[3] = (unsigned)((ga >> 32) & 0x01ffffffu)  // global_addr[56:32]
        | (2u << 30);                           // type=2 ("image")
  i32x8 g1;
  g1[0] = (int)(2u << 16);                      // data_size=4B; wg_mask=0
  g1[1] = (int)((tensor_k & 0xffffu) << 16);    // tensor_dim0[15:0] @63:48
  g1[2] = (int)(((tensor_k >> 16) & 0xffffu) |  // tensor_dim0[31:16]
                ((tensor_n & 0xffffu) << 16));  // tensor_dim1[15:0]
  g1[3] = (int)(((tensor_n >> 16) & 0xffffu) |  // tensor_dim1[31:16]
                ((tile_k & 0xffffu) << 16));    // tile_dim0 @127:112
  g1[4] = (int)(tile_n & 0xffffu);              // tile_dim1; tile_dim2=0
  g1[5] = (int)row_stride;                      // tensor_dim0_stride[31:0]
  g1[6] = 0;                                    // stride hi / dim1_stride
  g1[7] = 0;
  i32x4 z4 = {0,0,0,0};
#if defined(__clang_major__) && (__clang_major__ >= 23)
  i32x8 z8 = {0,0,0,0,0,0,0,0};
  __builtin_amdgcn_tensor_load_to_lds(g0, g1, z4, z4, z8, 0);
#else
  __builtin_amdgcn_tensor_load_to_lds(g0, g1, z4, z4, 0);
#endif
}

// ---------------- generic WMMA GEMM ----------------
// C[M,N](f32) = A[M,K](bf16, row-major, lda) * W[N,K](f32, row-major, ldw)^T
// Tile 128x64x32; 8 waves/WG, each wave owns 16 rows x all 64 cols:
// one A fragment -> 4 WMMAs per K-step. TDM double-buffers the fp32 W tile;
// 256 threads cooperatively repack it to bf16 pairs in LDS.
#define BM 128
#define BN 64
#define BK 32

__global__ __launch_bounds__(256) void k_gemm_bf16f32(
    const unsigned short* __restrict__ A, const float* __restrict__ W,
    float* __restrict__ C, int M, int N, int Kd, int lda, int ldw, int ldc) {
  __shared__ float        Wf[2][BN * BK];       // 2 x 8 KB fp32 tile (TDM dest)
  __shared__ unsigned int Wb[2][BN * BK / 2];   // 2 x 4 KB packed-bf16 tile
  const int bn = blockIdx.x * BN, bm = blockIdx.y * BM;
  const int tid = threadIdx.x, lane = tid & 31, wave = tid >> 5;
  const int nl = lane & 15, kh = lane >> 4;
  v8f acc[4] = {{}, {}, {}, {}};
  const int arow = bm + wave * 16 + nl;         // A-frag row (lanes 0-15 == 16-31)
  const int nsteps = Kd / BK;

  // prologue: TDM fetch of tile 0 (per-wave op; issue from wave 0 only)
  if (tid < 32)
    tdm_load_tile_f32((unsigned)(uintptr_t)(void*)Wf[0],
                      W + (size_t)bn * ldw,
                      BK, BN, (unsigned)Kd, (unsigned)N, (unsigned)ldw);

  for (int i = 0; i < nsteps; ++i) {
    const int cur = i & 1, nxt = cur ^ 1;
    if (tid < 32) __builtin_amdgcn_s_wait_tensorcnt(0);
    __syncthreads();                            // tile[cur] fp32 is in LDS

    // cooperative repack: 2048 fp32 -> 1024 bf16-pair dwords (8 elems/thread)
    {
      const float* src = Wf[cur];
      unsigned int* dst = Wb[cur];
#pragma unroll
      for (int j = 0; j < 4; ++j) {
        float lo = src[tid * 8 + 2 * j];
        float hi = src[tid * 8 + 2 * j + 1];
        dst[tid * 4 + j] = (unsigned)f2bf(lo) | ((unsigned)f2bf(hi) << 16);
      }
    }
    __syncthreads();                            // bf16 tile[cur] ready

    // overlap: kick TDM for tile[i+1] while we do the matrix math
    if (i + 1 < nsteps && tid < 32)
      tdm_load_tile_f32((unsigned)(uintptr_t)(void*)Wf[nxt],
                        W + (size_t)bn * ldw + (size_t)(i + 1) * BK,
                        BK, BN, (unsigned)Kd, (unsigned)N, (unsigned)ldw);

    // A fragment: lane(m,kh) holds K run [kh*16,+16).
    frag16 a;
    const unsigned short* ap = A + (size_t)arow * lda + i * BK;
    a.q[0] = *(const uint4*)(ap + kh * 16);
    a.q[1] = *(const uint4*)(ap + kh * 16 + 8);

    // 4 B fragments from packed-bf16 LDS (two ds_load_b128 each) + 4 WMMAs
#pragma unroll
    for (int t4 = 0; t4 < 4; ++t4) {
      const int n = t4 * 16 + nl;
      const uint4* bp = (const uint4*)&Wb[cur][n * (BK / 2) + kh * 8];
      frag16 b;
      b.q[0] = bp[0];
      b.q[1] = bp[1];
      acc[t4] = __builtin_amdgcn_wmma_f32_16x16x32_bf16(
          false, a.v, false, b.v, (short)0, acc[t4], false, false);
    }
    // no trailing barrier needed: next iteration's writes touch the other
    // buffer pair and are fenced by its two barriers.
  }

  // C layout: VGPR r, lanes 0-15 -> M=r, lanes 16-31 -> M=8+r; N=lane%16.
  const int mh = lane >> 4;
#pragma unroll
  for (int t4 = 0; t4 < 4; ++t4) {
#pragma unroll
    for (int r = 0; r < 8; ++r) {
      int row = bm + wave * 16 + r + 8 * mh;
      int col = bn + t4 * 16 + nl;
      if (row < M && col < N) C[(size_t)row * ldc + col] = acc[t4][r];
    }
  }
}

// ---------------- RMSNorm (row of HID) -> bf16 ----------------
__global__ __launch_bounds__(256) void k_rmsnorm(
    const float* __restrict__ x, const float* __restrict__ w,
    unsigned short* __restrict__ y) {
  __shared__ float red[256];
  const int s = blockIdx.x, tid = threadIdx.x;
  const float* row = x + (size_t)s * HID;
  float ss = 0.f;
  for (int i = tid; i < HID; i += 256) { float v = row[i]; ss += v * v; }
  red[tid] = ss; __syncthreads();
  for (int st = 128; st > 0; st >>= 1) {
    if (tid < st) red[tid] += red[tid + st];
    __syncthreads();
  }
  float inv = rsqrtf(red[0] / HID + EPSV);
  for (int i = tid; i < HID; i += 256)
    y[(size_t)s * HID + i] = f2bf(row[i] * inv * w[i]);
}

// ---------------- RoPE + q/k head-RMSNorm prep ----------------
// qkv f32 [S, KVH*(G+2)*D] -> qb bf16 [NH][S][D], kf f32 [KVH][S][D],
// vT f32 [KVH][D][S] (transposed so P@V fits the generic GEMM "W" side).
__global__ __launch_bounds__(64) void k_qkv_prep(
    const float* __restrict__ qkv, const float* __restrict__ cosp,
    const float* __restrict__ sinp, const float* __restrict__ qw,
    const float* __restrict__ kw, unsigned short* __restrict__ qb,
    float* __restrict__ kf, float* __restrict__ vT) {
  __shared__ float red[64];
  const int h = blockIdx.x / SEQ, s = blockIdx.x % SEQ, d = threadIdx.x;
  const float* base = qkv + (size_t)s * (KVHEADS * (GRP + 2) * HD)
                          + (size_t)h * (GRP + 2) * HD;
  const float c = cosp[s * HD + d], sn = sinp[s * HD + d];
  // k: rope + rmsnorm
  {
    float x  = base[GRP * HD + d];
    float xr = (d < HD / 2) ? -base[GRP * HD + d + HD / 2]
                            :  base[GRP * HD + d - HD / 2];
    float v = x * c + xr * sn;
    red[d] = v * v; __syncthreads();
    for (int st = 32; st > 0; st >>= 1) { if (d < st) red[d] += red[d + st]; __syncthreads(); }
    float inv = rsqrtf(red[0] / HD + EPSV);
    kf[((size_t)h * SEQ + s) * HD + d] = v * inv * kw[d];
    __syncthreads();
  }
  // v: transpose store
  vT[((size_t)h * HD + d) * SEQ + s] = base[(GRP + 1) * HD + d];
  // q heads: rope + rmsnorm -> bf16
  for (int g = 0; g < GRP; ++g) {
    float x  = base[g * HD + d];
    float xr = (d < HD / 2) ? -base[g * HD + d + HD / 2]
                            :  base[g * HD + d - HD / 2];
    float v = x * c + xr * sn;
    red[d] = v * v; __syncthreads();
    for (int st = 32; st > 0; st >>= 1) { if (d < st) red[d] += red[d + st]; __syncthreads(); }
    float inv = rsqrtf(red[0] / HD + EPSV);
    qb[((size_t)(h * GRP + g) * SEQ + s) * HD + d] = f2bf(v * inv * qw[d]);
    __syncthreads();
  }
}

// ---------------- causal softmax: scores f32 [S,S] -> P bf16 [S,S] --------
__global__ __launch_bounds__(256) void k_softmax_causal(
    const float* __restrict__ sc, unsigned short* __restrict__ P) {
  __shared__ float red[256];
  const int i = blockIdx.x, tid = threadIdx.x;
  const float* row = sc + (size_t)i * SEQ;
  const float scale = 0.125f;  // 1/sqrt(64)
  float mx = -3.4e38f;
  for (int j = tid; j <= i; j += 256) mx = fmaxf(mx, row[j] * scale);
  red[tid] = mx; __syncthreads();
  for (int st = 128; st > 0; st >>= 1) { if (tid < st) red[tid] = fmaxf(red[tid], red[tid + st]); __syncthreads(); }
  mx = red[0]; __syncthreads();
  float sum = 0.f;
  for (int j = tid; j <= i; j += 256) sum += __expf(row[j] * scale - mx);
  red[tid] = sum; __syncthreads();
  for (int st = 128; st > 0; st >>= 1) { if (tid < st) red[tid] += red[tid + st]; __syncthreads(); }
  float inv = 1.f / red[0];
  for (int j = tid; j < SEQ; j += 256)
    P[(size_t)i * SEQ + j] =
        (j <= i) ? f2bf(__expf(row[j] * scale - mx) * inv) : (unsigned short)0;
}

// ---------------- small elementwise kernels ----------------
__global__ void k_f2bf_buf(const float* __restrict__ in, unsigned short* __restrict__ out, int n) {
  int i = blockIdx.x * blockDim.x + threadIdx.x;
  if (i < n) out[i] = f2bf(in[i]);
}
__global__ void k_add2(const float* __restrict__ a, const float* __restrict__ b,
                       float* __restrict__ o, int n) {
  int i = blockIdx.x * blockDim.x + threadIdx.x;
  if (i < n) o[i] = a[i] + b[i];
}
__global__ void k_zero(float* __restrict__ o, int n) {
  int i = blockIdx.x * blockDim.x + threadIdx.x;
  if (i < n) o[i] = 0.f;
}
__global__ void k_final(const float* __restrict__ a, const float* __restrict__ b,
                        const float* __restrict__ c, float* __restrict__ o, int n) {
  int i = blockIdx.x * blockDim.x + threadIdx.x;
  if (i < n) o[i] = a[i] + b[i] + c[i];
}
// act = gu[:, :FF] * silu(gu[:, FF:]) -> bf16
__global__ void k_swiglu(const float* __restrict__ gu, unsigned short* __restrict__ act, int rows) {
  int idx = blockIdx.x * blockDim.x + threadIdx.x;
  int total = rows * FFD;
  if (idx >= total) return;
  int r = idx / FFD, f = idx % FFD;
  float x1 = gu[(size_t)r * 2 * FFD + f];
  float x2 = gu[(size_t)r * 2 * FFD + FFD + f];
  float sig = 1.f / (1.f + __expf(-x2));
  act[idx] = f2bf(x1 * (x2 * sig));
}

// ---------------- MoE routing ----------------
__global__ void k_topk(const float* __restrict__ logits, int* __restrict__ idx,
                       float* __restrict__ wgt) {
  int t = blockIdx.x * blockDim.x + threadIdx.x;
  if (t >= SEQ) return;
  float g[NEXP];
  const float* row = logits + (size_t)t * NEXP;
  float mx = row[0];
  for (int e = 1; e < NEXP; ++e) mx = fmaxf(mx, row[e]);
  float sum = 0.f;
  for (int e = 0; e < NEXP; ++e) { g[e] = __expf(row[e] - mx); sum += g[e]; }
  float inv = 1.f / sum;
  for (int e = 0; e < NEXP; ++e) g[e] *= inv;
  int chosen[TOPK];
  float tsum = 0.f;
  for (int k = 0; k < TOPK; ++k) {
    int bi = 0; float bv = -1.f;
    for (int e = 0; e < NEXP; ++e) {
      bool used = false;
      for (int kk = 0; kk < k; ++kk) used = used || (chosen[kk] == e);
      if (!used && g[e] > bv) { bv = g[e]; bi = e; }
    }
    chosen[k] = bi;
    idx[t * TOPK + k] = bi;
    wgt[t * TOPK + k] = bv;
    tsum += bv;
  }
  tsum = fmaxf(tsum, F32EPS);
  for (int k = 0; k < TOPK; ++k) wgt[t * TOPK + k] /= tsum;
}

// Reference priority order: flatten transpose(top_i) -> (k outer, s inner);
// one thread per expert scans in that order and assigns capacity slots.
__global__ void k_slots(const int* __restrict__ idx, const float* __restrict__ wgt,
                        int* __restrict__ slot_tok, float* __restrict__ slot_w,
                        int* __restrict__ cnt) {
  int e = threadIdx.x;
  if (e >= NEXP) return;
  int c = 0;
  for (int kk = 0; kk < TOPK; ++kk)
    for (int s = 0; s < SEQ; ++s)
      if (idx[s * TOPK + kk] == e) {
        if (c < CAPE) {
          slot_tok[e * CAPE + c] = s;
          slot_w[e * CAPE + c]   = wgt[s * TOPK + kk];
        }
        ++c;
      }
  cnt[e] = (c < CAPE) ? c : CAPE;
}

__global__ void k_dispatch(const unsigned short* __restrict__ x2b,
                           const int* __restrict__ slot_tok,
                           const int* __restrict__ cnt,
                           unsigned short* __restrict__ disp) {
  int e = blockIdx.x / CAPE, c = blockIdx.x % CAPE;
  unsigned short* dst = disp + (size_t)blockIdx.x * HID;
  if (c < cnt[e]) {
    const unsigned short* src = x2b + (size_t)slot_tok[blockIdx.x] * HID;
    for (int i = threadIdx.x; i < HID; i += blockDim.x) dst[i] = src[i];
  } else {
    for (int i = threadIdx.x; i < HID; i += blockDim.x) dst[i] = 0;
  }
}

// moe[tok] += w * eout[c]. Tokens are unique within one expert; experts run
// as sequential kernels on one stream, so plain += is race-free.
__global__ void k_combine(const float* __restrict__ eout,
                          const int* __restrict__ slot_tok,
                          const float* __restrict__ slot_w,
                          const int* __restrict__ cnt_e,
                          float* __restrict__ moe) {
  int c = blockIdx.x;
  if (c >= cnt_e[0]) return;
  int s = slot_tok[c];
  float w = slot_w[c];
  float* dst = moe + (size_t)s * HID;
  const float* src = eout + (size_t)c * HID;
  for (int i = threadIdx.x; i < HID; i += blockDim.x) dst[i] += w * src[i];
}

// ---------------- orchestration ----------------
extern "C" void kernel_launch(void* const* d_in, const int* in_sizes, int n_in,
                              void* d_out, int out_size, void* d_ws, size_t ws_size,
                              hipStream_t stream) {
  (void)in_sizes; (void)n_in; (void)out_size; (void)ws_size;
  const float* hs   = (const float*)d_in[0];   // hidden_states [S,H]
  const float* cosp = (const float*)d_in[1];   // [1,S,64]
  const float* sinp = (const float*)d_in[2];
  const float* iln  = (const float*)d_in[3];
  const float* pln  = (const float*)d_in[4];
  const float* qkvw = (const float*)d_in[5];   // [2048,768]
  const float* ow   = (const float*)d_in[6];   // [768,1024]
  const float* qlnw = (const float*)d_in[7];
  const float* klnw = (const float*)d_in[8];
  const float* gw   = (const float*)d_in[9];   // [64,768]
  const float* sguw = (const float*)d_in[10];  // [6144,768]
  const float* sdw  = (const float*)d_in[11];  // [768,3072]
  const float* eguw = (const float*)d_in[12];  // [64,6144,768]
  const float* edw  = (const float*)d_in[13];  // [64,768,3072]
  float* out = (float*)d_out;

  // workspace carve-up (~100 MB; every region rewritten each call)
  char* wsb = (char*)d_ws;
  size_t off = 0;
  auto alloc = [&](size_t bytes) -> void* {
    void* p = wsb + off;
    off += (bytes + 255) & ~(size_t)255;
    return p;
  };
  unsigned short* xb    = (unsigned short*)alloc((size_t)SEQ * HID * 2);
  float*          qkvf  = (float*)alloc((size_t)SEQ * 2048 * 4);
  unsigned short* qb    = (unsigned short*)alloc((size_t)NHEADS * SEQ * HD * 2);
  float*          kf    = (float*)alloc((size_t)KVHEADS * SEQ * HD * 4);
  float*          vT    = (float*)alloc((size_t)KVHEADS * HD * SEQ * 4);
  float*          scores= (float*)alloc((size_t)SEQ * SEQ * 4);
  unsigned short* Pb    = (unsigned short*)alloc((size_t)SEQ * SEQ * 2);
  float*          attnf = (float*)alloc((size_t)SEQ * HQD * 4);
  unsigned short* attnb = (unsigned short*)alloc((size_t)SEQ * HQD * 2);
  float*          oout  = (float*)alloc((size_t)SEQ * HID * 4);
  float*          hidden= (float*)alloc((size_t)SEQ * HID * 4);
  unsigned short* x2b   = (unsigned short*)alloc((size_t)SEQ * HID * 2);
  float*          sgu   = (float*)alloc((size_t)SEQ * 2 * FFD * 4);
  unsigned short* sact  = (unsigned short*)alloc((size_t)SEQ * FFD * 2);
  float*          sout  = (float*)alloc((size_t)SEQ * HID * 4);
  float*          logits= (float*)alloc((size_t)SEQ * NEXP * 4);
  int*            tki   = (int*)alloc((size_t)SEQ * TOPK * 4);
  float*          tkw   = (float*)alloc((size_t)SEQ * TOPK * 4);
  int*            slot_tok = (int*)alloc((size_t)NEXP * CAPE * 4);
  float*          slot_w   = (float*)alloc((size_t)NEXP * CAPE * 4);
  int*            cnt   = (int*)alloc((size_t)NEXP * 4);
  unsigned short* disp  = (unsigned short*)alloc((size_t)NEXP * CAPE * HID * 2);
  float*          egu   = (float*)alloc((size_t)CAPE * 2 * FFD * 4);
  unsigned short* eact  = (unsigned short*)alloc((size_t)CAPE * FFD * 2);
  float*          eoutb = (float*)alloc((size_t)CAPE * HID * 4);
  float*          moe   = (float*)alloc((size_t)SEQ * HID * 4);

  const int NE = SEQ * HID;

  // 1) input RMSNorm -> bf16
  k_rmsnorm<<<SEQ, 256, 0, stream>>>(hs, iln, xb);
  // 2) QKV projection: [S,768] x [2048,768]^T
  k_gemm_bf16f32<<<dim3(2048 / BN, SEQ / BM), 256, 0, stream>>>(
      xb, qkvw, qkvf, SEQ, 2048, HID, HID, HID, 2048);
  // 3) RoPE + q/k per-head RMSNorm, V transpose
  k_qkv_prep<<<KVHEADS * SEQ, 64, 0, stream>>>(qkvf, cosp, sinp, qlnw, klnw, qb, kf, vT);
  // 4) attention per head: scores -> causal softmax -> P @ V
  for (int h = 0; h < NHEADS; ++h) {
    const unsigned short* qh = qb + (size_t)h * SEQ * HD;
    const float* kh = kf + (size_t)(h / GRP) * SEQ * HD;
    const float* vh = vT + (size_t)(h / GRP) * HD * SEQ;
    k_gemm_bf16f32<<<dim3(SEQ / BN, SEQ / BM), 256, 0, stream>>>(
        qh, kh, scores, SEQ, SEQ, HD, HD, HD, SEQ);
    k_softmax_causal<<<SEQ, 256, 0, stream>>>(scores, Pb);
    k_gemm_bf16f32<<<dim3(HD / BN, SEQ / BM), 256, 0, stream>>>(
        Pb, vh, attnf + h * HD, SEQ, HD, SEQ, SEQ, SEQ, HQD);
  }
  // 5) O projection
  k_f2bf_buf<<<CDIV(SEQ * HQD, 256), 256, 0, stream>>>(attnf, attnb, SEQ * HQD);
  k_gemm_bf16f32<<<dim3(HID / BN, SEQ / BM), 256, 0, stream>>>(
      attnb, ow, oout, SEQ, HID, HQD, HQD, HQD, HID);
  // 6) residual + post RMSNorm
  k_add2<<<CDIV(NE, 256), 256, 0, stream>>>(hs, oout, hidden, NE);
  k_rmsnorm<<<SEQ, 256, 0, stream>>>(hidden, pln, x2b);
  // 7) shared SwiGLU MLP
  k_gemm_bf16f32<<<dim3(2 * FFD / BN, SEQ / BM), 256, 0, stream>>>(
      x2b, sguw, sgu, SEQ, 2 * FFD, HID, HID, HID, 2 * FFD);
  k_swiglu<<<CDIV(SEQ * FFD, 256), 256, 0, stream>>>(sgu, sact, SEQ);
  k_gemm_bf16f32<<<dim3(HID / BN, SEQ / BM), 256, 0, stream>>>(
      sact, sdw, sout, SEQ, HID, FFD, FFD, FFD, HID);
  // 8) router: logits -> top-k -> capacity slots -> dispatch
  k_gemm_bf16f32<<<dim3(NEXP / BN, SEQ / BM), 256, 0, stream>>>(
      x2b, gw, logits, SEQ, NEXP, HID, HID, HID, NEXP);
  k_topk<<<CDIV(SEQ, 256), 256, 0, stream>>>(logits, tki, tkw);
  k_slots<<<1, NEXP, 0, stream>>>(tki, tkw, slot_tok, slot_w, cnt);
  k_dispatch<<<NEXP * CAPE, 256, 0, stream>>>(x2b, slot_tok, cnt, disp);
  k_zero<<<CDIV(NE, 256), 256, 0, stream>>>(moe, NE);
  // 9) experts (sequential on the stream; weights stream HBM->LDS via TDM)
  for (int e = 0; e < NEXP; ++e) {
    const unsigned short* de = disp + (size_t)e * CAPE * HID;
    const float* gue = eguw + (size_t)e * 2 * FFD * HID;
    const float* dwe = edw + (size_t)e * HID * FFD;
    k_gemm_bf16f32<<<dim3(2 * FFD / BN, CAPE / BM), 256, 0, stream>>>(
        de, gue, egu, CAPE, 2 * FFD, HID, HID, HID, 2 * FFD);
    k_swiglu<<<CDIV(CAPE * FFD, 256), 256, 0, stream>>>(egu, eact, CAPE);
    k_gemm_bf16f32<<<dim3(HID / BN, CAPE / BM), 256, 0, stream>>>(
        eact, dwe, eoutb, CAPE, HID, FFD, FFD, FFD, HID);
    k_combine<<<CAPE, 256, 0, stream>>>(eoutb, slot_tok + e * CAPE,
                                        slot_w + e * CAPE, cnt + e, moe);
  }
  // 10) out = residual2 + shared + combined
  k_final<<<CDIV(NE, 256), 256, 0, stream>>>(hidden, sout, moe, out, NE);
}
